// ConceptAttentionProto_28449863369189
// MI455X (gfx1250) — matrix operand order repaired and checked by reading, not compiled, not run on hardware
//
#include <hip/hip_runtime.h>

typedef __fp16   v2fp __attribute__((ext_vector_type(2)));
typedef _Float16 v8h  __attribute__((ext_vector_type(8)));
typedef _Float16 v16h __attribute__((ext_vector_type(16)));
typedef float    v8f  __attribute__((ext_vector_type(8)));

#define FEAT   128
#define LQ     1024
#define NBATCH 8
#define NQ     (NBATCH * LQ)      // 8192 queries
#define NSLOT  (64 * 256)         // 16384 pool slots

// ---------------- preprocessing: f16 copies in workspace ----------------

// q_h[(n*LQ + l)*FEAT + c] = x[n][c][l] * (1/sqrt(128)), as f16
__global__ void ca_prep_q(const float* __restrict__ x, _Float16* __restrict__ qh) {
    int idx = blockIdx.x * blockDim.x + threadIdx.x;          // over NQ*FEAT
    if (idx >= NQ * FEAT) return;
    int c = idx & (FEAT - 1);
    int q = idx >> 7;
    int l = q & (LQ - 1);
    int n = q >> 10;
    const float scale = 0.088388347648318447f;                // 1/sqrt(128)
    qh[idx] = (_Float16)(x[(size_t)n * FEAT * LQ + (size_t)c * LQ + l] * scale);
}

// pool_h (row major, [slot][feat]) and pool_t (feat major, [feat][slot])
__global__ void ca_prep_pool(const float* __restrict__ p,
                             _Float16* __restrict__ ph,
                             _Float16* __restrict__ pt) {
    int idx = blockIdx.x * blockDim.x + threadIdx.x;          // over NSLOT*FEAT
    if (idx >= NSLOT * FEAT) return;
    int c = idx & (FEAT - 1);
    int s = idx >> 7;
    _Float16 v = (_Float16)p[idx];
    ph[idx] = v;
    pt[(size_t)c * NSLOT + s] = v;
}

// ---------------- helpers ----------------

// pack two f32 -> packed 2xf16 (v_cvt_pk_rtz_f16_f32), as 32-bit int
static __device__ inline int pk2h(float a, float b) {
    v2fp h = __builtin_amdgcn_cvt_pkrtz(a, b);
    int r;
    __builtin_memcpy(&r, &h, 4);
    return r;
}

// Build one P^T B-fragment (K=0..31 over two 16-row score C-tiles) from
// exp'ed scores pa (rows +0..15) / pb (rows +16..31), using packed half2
// values and a single lane^16 exchange per pair.
static __device__ inline void build_bp(const float* pa, const float* pb,
                                       bool hi, int* bp) {
#pragma unroll
    for (int v = 0; v < 4; ++v) {
        int h0 = pk2h(pa[2 * v], pa[2 * v + 1]);   // rows 2v,2v+1 (lane half)
        int h1 = pk2h(pb[2 * v], pb[2 * v + 1]);
        int s0 = __shfl_xor(h0, 16, 32);
        int s1 = __shfl_xor(h1, 16, 32);
        bp[v]     = hi ? s1 : h0;                  // K = 2v,2v+1   | 16+2v,17+2v
        bp[4 + v] = hi ? h1 : s0;                  // K = 8+2v,9+2v | 24+2v,25+2v
    }
}

// ---------------- flash-attention main kernel ----------------
// One wave owns 16 queries; streams all 16384 pool rows in chunks of 64 with
// the transposed formulation:
//   T (64x16)   = K_tile * Q^T      (A = pool rows, B = Q^T; contiguous loads)
//   O^T(128x16) += V^T * P^T        (A = pool_t rows, B = exp(T) via shuffles)
// Online softmax state (running max M, running sum L) is per-lane scalar.

__global__ __launch_bounds__(128, 1)
void ca_flash_attn(const _Float16* __restrict__ qh,
                   const _Float16* __restrict__ ph,
                   const _Float16* __restrict__ pt,
                   float* __restrict__ out) {
    const int lane = threadIdx.x & 31;
    const int wave = threadIdx.x >> 5;
    const int tile = blockIdx.x * 4 + wave;                   // 0..511
    const int q0   = tile * 16;
    const int col  = lane & 15;                               // query column
    const bool hi  = lane >= 16;

    // ---- Q^T B-fragments, one per 32-wide K chunk (loaded once) ----
    v16h Bq[4];
    {
        const _Float16* qp = qh + (size_t)(q0 + col) * FEAT + (hi ? 8 : 0);
#pragma unroll
        for (int ch = 0; ch < 4; ++ch) {
            const _Float16* s = qp + ch * 32;
            v16h b;
            *((v8h*)&b + 0) = *(const v8h*)(s);
            *((v8h*)&b + 1) = *(const v8h*)(s + 16);
            Bq[ch] = b;
        }
    }

    // ---- accumulators ----
    v8f O[8];
#pragma unroll
    for (int f = 0; f < 8; ++f)
        O[f] = (v8f){0.f, 0.f, 0.f, 0.f, 0.f, 0.f, 0.f, 0.f};
    float M    = -1e30f;
    float Lsum = 0.0f;

    for (int kv = 0; kv < NSLOT; kv += 64) {
        // prefetch next KV tile (speculative; dropped if OOB)
        __builtin_prefetch(ph + (size_t)(kv + 64 + col) * FEAT, 0, 3);
        __builtin_prefetch(pt + (size_t)col * NSLOT + kv + 64, 0, 3);

        // ===== T = K_tile * Q^T : four 16x16 C-tiles (rows kv+16t..kv+16t+15)
        v8f T[4];
#pragma unroll
        for (int t = 0; t < 4; ++t)
            T[t] = (v8f){0.f, 0.f, 0.f, 0.f, 0.f, 0.f, 0.f, 0.f};
#pragma unroll
        for (int ch = 0; ch < 4; ++ch) {
            const _Float16* a0 =
                ph + (size_t)(kv + col) * FEAT + ch * 32 + (hi ? 8 : 0);
#pragma unroll
            for (int t = 0; t < 4; ++t) {
                const _Float16* a = a0 + (size_t)(t * 16) * FEAT;
                v16h A;
                *((v8h*)&A + 0) = *(const v8h*)(a);
                *((v8h*)&A + 1) = *(const v8h*)(a + 16);
                T[t] = __builtin_amdgcn_wmma_f32_16x16x32_f16(
                    false, A, false, Bq[ch], (short)0, T[t], false, false);
            }
        }

        // ===== online softmax update (per-lane scalar state) =====
        float tm = T[0][0];
#pragma unroll
        for (int t = 0; t < 4; ++t)
#pragma unroll
            for (int r = 0; r < 8; ++r) tm = fmaxf(tm, T[t][r]);
        tm = fmaxf(tm, __shfl_xor(tm, 16, 32));
        float newM  = fmaxf(M, tm);
        float alpha = __expf(M - newM);
        M = newM;

        float rs = 0.f;
        int bp01[8], bp23[8];
        {
            float pa[8], pb[8];
#pragma unroll
            for (int r = 0; r < 8; ++r) {
                pa[r] = __expf(T[0][r] - M); rs += pa[r];
                pb[r] = __expf(T[1][r] - M); rs += pb[r];
            }
            build_bp(pa, pb, hi, bp01);
        }
        {
            float pa[8], pb[8];
#pragma unroll
            for (int r = 0; r < 8; ++r) {
                pa[r] = __expf(T[2][r] - M); rs += pa[r];
                pb[r] = __expf(T[3][r] - M); rs += pb[r];
            }
            build_bp(pa, pb, hi, bp23);
        }
        rs += __shfl_xor(rs, 16, 32);
        Lsum = Lsum * alpha + rs;

#pragma unroll
        for (int f = 0; f < 8; ++f)
#pragma unroll
            for (int r = 0; r < 8; ++r) O[f][r] *= alpha;

        v16h Bp0, Bp1;
        __builtin_memcpy(&Bp0, bp01, 32);
        __builtin_memcpy(&Bp1, bp23, 32);

        // ===== O^T += V^T * P^T : 8 feature blocks x 2 K-chunks =====
#pragma unroll
        for (int f = 0; f < 8; ++f) {
            const _Float16* vp0 =
                pt + (size_t)(f * 16 + col) * NSLOT + kv + (hi ? 16 : 0);
            v16h Av0, Av1;
            *((v8h*)&Av0 + 0) = *(const v8h*)(vp0);
            *((v8h*)&Av0 + 1) = *(const v8h*)(vp0 + 8);
            *((v8h*)&Av1 + 0) = *(const v8h*)(vp0 + 32);
            *((v8h*)&Av1 + 1) = *(const v8h*)(vp0 + 40);
            O[f] = __builtin_amdgcn_wmma_f32_16x16x32_f16(
                false, Av0, false, Bp0, (short)0, O[f], false, false);
            O[f] = __builtin_amdgcn_wmma_f32_16x16x32_f16(
                false, Av1, false, Bp1, (short)0, O[f], false, false);
        }
    }

    // ===== epilogue: normalize and scatter to out[n][c][l] =====
    float inv = 1.0f / Lsum;
    int q = q0 + col;
    int l = q & (LQ - 1);
    int n = q >> 10;
    float* ob = out + (size_t)n * FEAT * LQ + l;
#pragma unroll
    for (int f = 0; f < 8; ++f) {
#pragma unroll
        for (int r = 0; r < 8; ++r) {
            int c = f * 16 + r + (hi ? 8 : 0);
            ob[(size_t)c * LQ] = O[f][r] * inv;
        }
    }
}

// ---------------- launcher ----------------
extern "C" void kernel_launch(void* const* d_in, const int* in_sizes, int n_in,
                              void* d_out, int out_size, void* d_ws, size_t ws_size,
                              hipStream_t stream) {
    const float* x    = (const float*)d_in[0];   // [8,128,32,32]
    const float* pool = (const float*)d_in[1];   // [16384,128]
    float* out        = (float*)d_out;           // [8,128,32,32]

    _Float16* qh = (_Float16*)d_ws;                          // 2 MB
    _Float16* ph = qh + (size_t)NQ * FEAT;                   // 4 MB
    _Float16* pt = ph + (size_t)NSLOT * FEAT;                // 4 MB

    {
        int total = NQ * FEAT;
        ca_prep_q<<<(total + 255) / 256, 256, 0, stream>>>(x, qh);
    }
    {
        int total = NSLOT * FEAT;
        ca_prep_pool<<<(total + 255) / 256, 256, 0, stream>>>(pool, ph, pt);
    }
    // 512 query tiles of 16, 4 waves (tiles) per 128-thread block
    ca_flash_attn<<<128, 128, 0, stream>>>(qh, ph, pt, out);
}
